// DiseaseModel_80539226735118
// MI455X (gfx1250) — compile-verified, matrix operand order checked by reading, not run on hardware
//
#include <hip/hip_runtime.h>

#define VSZ   20000
#define LSEQ  128
#define DIM   256
#define NDEMO 70
#define HIDN  1024
#define NSAMP 2048
#define ESTR  264        // padded LDS row stride in halves (256 + 8)
#define XDIM  (NDEMO + DIM)   // 326
#define XPAD  352             // XDIM padded to a multiple of 32 (11 K-steps)

typedef __attribute__((ext_vector_type(16))) _Float16 v16h;
typedef __attribute__((ext_vector_type(8)))  _Float16 v8h;
typedef __attribute__((ext_vector_type(8)))  float    v8f;

__device__ __forceinline__ v8f wmma16x16x32(v16h a, v16h b, v8f c) {
  // D = A(16x32 f16) * B(32x16 f16) + C(16x16 f32)
  return __builtin_amdgcn_wmma_f32_16x16x32_f16(false, a, false, b, (short)0, c,
                                                false, false);
}

// Load a 16x32 f16 fragment from an LDS matrix (row-major, `stride` halves).
// CDNA5 16-bit A/B layout (CDNA5 ISA): row = base + (lane&15), lane-half hi
// selects K-subblock; elements 0..7 <- [kbase+8*hi ..], 8..15 <- [kbase+16+8*hi ..].
__device__ __forceinline__ v16h load_frag_lds(const _Float16* base, int rowbase,
                                              int kbase, int lane, int stride) {
  int r  = rowbase + (lane & 15);
  int hi = lane >> 4;
  const _Float16* p = base + r * stride + kbase + 8 * hi;
  v8h lo = *(const v8h*)(p);
  v8h h2 = *(const v8h*)(p + 16);
  v16h a;
#pragma unroll
  for (int i = 0; i < 8; ++i) { a[i] = lo[i]; a[i + 8] = h2[i]; }
  return a;
}

__device__ __forceinline__ void wait_asynccnt0() {
#if __has_builtin(__builtin_amdgcn_s_wait_asynccnt)
  __builtin_amdgcn_s_wait_asynccnt(0);
#else
  asm volatile("s_wait_asynccnt 0" ::: "memory");
#endif
}

// ---------------- prep: M = (Wq @ Wk^T) / sqrt(D), then pack to B-frag order --
__global__ __launch_bounds__(256) void prep_m(const float* __restrict__ Wq,
                                              const float* __restrict__ Wk,
                                              float* __restrict__ Mtmp) {
  int r = blockIdx.x, c = threadIdx.x;
  float acc = 0.f;
  for (int d = 0; d < DIM; ++d) acc += Wq[r * DIM + d] * Wk[c * DIM + d];
  Mtmp[r * DIM + c] = acc * 0.0625f;  // fold 1/sqrt(256)
}

__global__ __launch_bounds__(256) void pack_m(const float* __restrict__ Mtmp,
                                              _Float16* __restrict__ Mpk) {
  int b = blockIdx.x;            // b = kt*16 + nt
  int kt = b >> 4, nt = b & 15;
  for (int e = threadIdx.x; e < 512; e += 256) {
    int l = e >> 4, j = e & 15;
    int hi = l >> 4;
    int col = nt * 16 + (l & 15);
    int offK = (j < 8) ? (8 * hi + j) : (16 + 8 * hi + (j - 8));
    Mpk[b * 512 + e] = (_Float16)Mtmp[(kt * 32 + offK) * DIM + col];
  }
}

// pack W1 (326x1024 f32) into f16 B-fragment order, zero-padding K to 352
__global__ __launch_bounds__(256) void pack_w1(const float* __restrict__ W1,
                                               _Float16* __restrict__ W1pk) {
  int b = blockIdx.x;            // b = kt*64 + nt   (kt<11, nt<64)
  int kt = b >> 6, nt = b & 63;
  for (int e = threadIdx.x; e < 512; e += 256) {
    int l = e >> 4, j = e & 15;
    int hi = l >> 4;
    int col = nt * 16 + (l & 15);
    int offK = (j < 8) ? (8 * hi + j) : (16 + 8 * hi + (j - 8));
    int k = kt * 32 + offK;
    float v = (k < XDIM) ? W1[k * HIDN + col] : 0.f;
    W1pk[b * 512 + e] = (_Float16)v;
  }
}

// ---------------- fused per-sample attention kernel -----------------------
__global__ __launch_bounds__(256) void fused_sample(
    const float* __restrict__ E, const _Float16* __restrict__ Mpk,
    const float* __restrict__ Wv, const float* __restrict__ demo,
    const float* __restrict__ hist_times, const float* __restrict__ event_time,
    const int* __restrict__ codes, const int* __restrict__ lengths,
    const float* __restrict__ c_attn, const float* __restrict__ c_pool,
    _Float16* __restrict__ Xg) {
  __shared__ __align__(16) _Float16 emb[LSEQ * ESTR];    // 66 KB
  __shared__ __align__(16) _Float16 tmat[LSEQ * ESTR];   // 66 KB
  __shared__ float times_s[LSEQ];
  __shared__ float w_s[LSEQ];
  __shared__ float u_s[LSEQ];
  __shared__ float upart[8 * LSEQ];
  __shared__ float y_s[DIM];
  __shared__ float x_s[XDIM + 2];
  __shared__ float red[256];
  __shared__ int   codes_s[LSEQ];

  const int n = blockIdx.x;
  const int tid = threadIdx.x;
  const int lane = tid & 31, wave = tid >> 5;
  const int len = lengths[n];
  const float ca = c_attn[0], cp = c_pool[0];
  const float et = event_time[n];

  if (tid < LSEQ) {
    codes_s[tid] = codes[n * LSEQ + tid];
    times_s[tid] = hist_times[n * LSEQ + tid];
  }
  __syncthreads();

  // ---- gather E rows, f32 -> f16, 16B global loads ----
  for (int e4 = tid; e4 < LSEQ * DIM / 4; e4 += 256) {
    int row = e4 >> 6;
    int c4 = (e4 & 63) << 2;
    float4 v = *(const float4*)(E + (size_t)codes_s[row] * DIM + c4);
    _Float16* dst = &emb[row * ESTR + c4];
    dst[0] = (_Float16)v.x; dst[1] = (_Float16)v.y;
    dst[2] = (_Float16)v.z; dst[3] = (_Float16)v.w;
  }

  // ---- pooling weights w = softmax(-c_pool*(event_time - t), masked) ----
  float wval = (tid < LSEQ && tid < len) ? (-cp * (et - times_s[tid])) : -3.0e38f;
  red[tid] = wval;
  __syncthreads();
  for (int s = 128; s > 0; s >>= 1) {
    if (tid < s) red[tid] = fmaxf(red[tid], red[tid + s]);
    __syncthreads();
  }
  float wmax = red[0];
  __syncthreads();
  float pe = __expf(wval - wmax);        // masked lanes underflow to 0
  red[tid] = pe;
  __syncthreads();
  for (int s = 128; s > 0; s >>= 1) {
    if (tid < s) red[tid] += red[tid + s];
    __syncthreads();
  }
  float winv = 1.f / red[0];
  __syncthreads();
  if (tid < LSEQ) w_s[tid] = pe * winv;
  __syncthreads();  // emb + w ready

  // ---- S1: t = emb @ M  (wave owns M-tile `wave`; 16 N-tiles x 8 K-steps) ----
  for (int nt = 0; nt < 16; ++nt) {
    v8f c = {0.f, 0.f, 0.f, 0.f, 0.f, 0.f, 0.f, 0.f};
#pragma unroll
    for (int kt = 0; kt < 8; ++kt) {
      v16h a = load_frag_lds(emb, 16 * wave, kt * 32, lane, ESTR);
      v16h b = *(const v16h*)(Mpk + ((size_t)((kt * 16 + nt) * 32 + lane)) * 16);
      c = wmma16x16x32(a, b, c);
    }
    int hi = lane >> 4, col = nt * 16 + (lane & 15);
#pragma unroll
    for (int p = 0; p < 8; ++p)
      tmat[(16 * wave + p + 8 * hi) * ESTR + col] = (_Float16)c[p];
  }
  __syncthreads();

  // ---- S2: scores = t @ emb^T, kept entirely in accumulator VGPRs ----
  v8f acc[8];
#pragma unroll
  for (int nt = 0; nt < 8; ++nt) acc[nt] = (v8f){0.f,0.f,0.f,0.f,0.f,0.f,0.f,0.f};
#pragma unroll
  for (int kt = 0; kt < 8; ++kt) {
    v16h a = load_frag_lds(tmat, 16 * wave, kt * 32, lane, ESTR);
#pragma unroll
    for (int nt = 0; nt < 8; ++nt) {
      v16h b = load_frag_lds(emb, nt * 16, kt * 32, lane, ESTR);  // emb^T as B
      acc[nt] = wmma16x16x32(a, b, acc[nt]);
    }
  }

  // ---- time decay + mask + row softmax in registers ----
  const int hi = lane >> 4, jl = lane & 15;
  float trow[8], wrow[8], rmax[8], rsum[8];
#pragma unroll
  for (int p = 0; p < 8; ++p) {
    int i = 16 * wave + p + 8 * hi;
    trow[p] = times_s[i];
    wrow[p] = w_s[i];
    rmax[p] = -3.0e38f;
    rsum[p] = 0.f;
  }
#pragma unroll
  for (int nt = 0; nt < 8; ++nt) {
    int j = nt * 16 + jl;
    float tj = times_s[j];
    bool ok = (j < len);
#pragma unroll
    for (int p = 0; p < 8; ++p) {
      float s = acc[nt][p] - ca * fabsf(trow[p] - tj);
      s = ok ? s : -3.0e38f;
      acc[nt][p] = s;
      rmax[p] = fmaxf(rmax[p], s);
    }
  }
#pragma unroll
  for (int p = 0; p < 8; ++p)
    for (int off = 1; off < 16; off <<= 1)
      rmax[p] = fmaxf(rmax[p], __shfl_xor(rmax[p], off, 32));
#pragma unroll
  for (int nt = 0; nt < 8; ++nt)
#pragma unroll
    for (int p = 0; p < 8; ++p) {
      float e = __expf(acc[nt][p] - rmax[p]);
      acc[nt][p] = e;
      rsum[p] += e;
    }
#pragma unroll
  for (int p = 0; p < 8; ++p) {
    for (int off = 1; off < 16; off <<= 1)
      rsum[p] += __shfl_xor(rsum[p], off, 32);
    rsum[p] = 1.f / rsum[p];
  }
  // u[j] contributions: sum_i w[i]*attn[i][j]
#pragma unroll
  for (int nt = 0; nt < 8; ++nt) {
    float s = 0.f;
#pragma unroll
    for (int p = 0; p < 8; ++p) s += wrow[p] * acc[nt][p] * rsum[p];
    s += __shfl_xor(s, 16, 32);          // combine the two lane halves
    if (lane < 16) upart[wave * LSEQ + nt * 16 + lane] = s;
  }
  __syncthreads();

  if (tid < LSEQ) {
    float s = 0.f;
#pragma unroll
    for (int w = 0; w < 8; ++w) s += upart[w * LSEQ + tid];
    u_s[tid] = s;
  }
  __syncthreads();

  // ---- y = u @ emb ; repr = y @ Wv ; x = [demo, repr] -> Xg (f16) ----
  {
    float yd = 0.f;
    for (int j = 0; j < LSEQ; ++j) yd += u_s[j] * (float)emb[j * ESTR + tid];
    y_s[tid] = yd;
  }
  __syncthreads();
  if (tid < NDEMO) x_s[tid] = demo[n * NDEMO + tid];
  {
    float re = 0.f;
    for (int d = 0; d < DIM; ++d) re += y_s[d] * Wv[d * DIM + tid];
    x_s[NDEMO + tid] = re;
  }
  __syncthreads();
  for (int i = tid; i < XPAD; i += 256)
    Xg[(size_t)n * XPAD + i] = (i < XDIM) ? (_Float16)x_s[i] : (_Float16)0.f;
}

// ---------------- batched MLP + loss: WMMA over X(2048x352)@W1(352x1024) ---
__global__ __launch_bounds__(256) void mlp_loss(
    const _Float16* __restrict__ Xg, const _Float16* __restrict__ W1pk,
    const float* __restrict__ b1, const float* __restrict__ W2,
    const float* __restrict__ b2, const float* __restrict__ labels,
    float* __restrict__ loss_ws) {
  __shared__ __align__(16) _Float16 xtile[16 * XPAD];  // 11264 B, contiguous rows
  __shared__ float rowsum[8 * 16];

  const int tid = threadIdx.x;
  const int lane = tid & 31, wave = tid >> 5;
  const int m0 = blockIdx.x * 16;

  // ---- stage 16 X rows via CDNA5 async global->LDS copies (ASYNCcnt) ----
  {
    const unsigned lds_base = (unsigned)(unsigned long long)(void*)&xtile[0];
    const unsigned long long gbase =
        (unsigned long long)(const void*)(Xg + (size_t)m0 * XPAD);
    for (int e = tid; e < (16 * XPAD * 2) / 16; e += 256) {  // 704 x 16B chunks
      unsigned dst = lds_base + e * 16;
      unsigned long long src = gbase + (unsigned long long)e * 16;
      asm volatile("global_load_async_to_lds_b128 %0, %1, off"
                   :: "v"(dst), "v"(src) : "memory");
    }
  }
  wait_asynccnt0();
  __syncthreads();

  // ---- H-tile GEMM: wave owns 8 N-tiles; K = 11 steps of 32 ----
  const int hi = lane >> 4, jl = lane & 15;
  float rp[8];
#pragma unroll
  for (int p = 0; p < 8; ++p) rp[p] = 0.f;

  for (int nt = 0; nt < 8; ++nt) {
    const int ntg = wave * 8 + nt;       // global N-tile 0..63
    v8f c = {0.f, 0.f, 0.f, 0.f, 0.f, 0.f, 0.f, 0.f};
#pragma unroll
    for (int kt = 0; kt < 11; ++kt) {
      v16h a = load_frag_lds(xtile, 0, kt * 32, lane, XPAD);
      v16h b = *(const v16h*)(W1pk + ((size_t)((kt * 64 + ntg) * 32 + lane)) * 16);
      c = wmma16x16x32(a, b, c);
    }
    int col = ntg * 16 + jl;
    float bb = b1[col], w2 = W2[col];
#pragma unroll
    for (int p = 0; p < 8; ++p) {
      float h = fmaxf(c[p] + bb, 0.f);   // relu(x@W1 + b1)
      rp[p] += h * w2;                   // fold dot with W2
    }
  }
  // reduce over the 16 lanes of each half -> per-row partial dot
#pragma unroll
  for (int p = 0; p < 8; ++p)
    for (int off = 1; off < 16; off <<= 1)
      rp[p] += __shfl_xor(rp[p], off, 32);
  if (jl == 0) {
#pragma unroll
    for (int p = 0; p < 8; ++p) rowsum[wave * 16 + p + 8 * hi] = rp[p];
  }
  __syncthreads();

  if (tid < 16) {
    float s = 0.f;
#pragma unroll
    for (int w = 0; w < 8; ++w) s += rowsum[w * 16 + tid];
    float logit = s + b2[0];
    float lab = labels[m0 + tid];
    float lsp = fminf(logit, 0.f) - log1pf(__expf(-fabsf(logit)));
    float lsn = fminf(-logit, 0.f) - log1pf(__expf(-fabsf(logit)));
    loss_ws[m0 + tid] = -(2.f * lab * lsp + (1.f - lab) * lsn);
  }
}

// ---------------- final reduction ----------------------------------------
__global__ __launch_bounds__(256) void reduce_loss(
    const float* __restrict__ loss_ws, const float* __restrict__ c_attn,
    const float* __restrict__ c_pool, float* __restrict__ out) {
  __shared__ float red[256];
  int tid = threadIdx.x;
  float s = 0.f;
  for (int i = tid; i < NSAMP; i += 256) s += loss_ws[i];
  red[tid] = s;
  __syncthreads();
  for (int k = 128; k > 0; k >>= 1) {
    if (tid < k) red[tid] += red[tid + k];
    __syncthreads();
  }
  if (tid == 0) {
    float ca = c_attn[0], cp = c_pool[0];
    out[0] = red[0] * (1.f / (float)NSAMP) + ca * ca + cp * cp;
  }
}

extern "C" void kernel_launch(void* const* d_in, const int* in_sizes, int n_in,
                              void* d_out, int out_size, void* d_ws, size_t ws_size,
                              hipStream_t stream) {
  const float* E      = (const float*)d_in[0];
  const float* Wq     = (const float*)d_in[1];
  const float* Wk     = (const float*)d_in[2];
  const float* Wv     = (const float*)d_in[3];
  const float* c_attn = (const float*)d_in[4];
  const float* c_pool = (const float*)d_in[5];
  const float* W1     = (const float*)d_in[6];
  const float* b1     = (const float*)d_in[7];
  const float* W2     = (const float*)d_in[8];
  const float* b2     = (const float*)d_in[9];
  const float* demo   = (const float*)d_in[10];
  const float* htime  = (const float*)d_in[11];
  const float* etime  = (const float*)d_in[12];
  const float* labels = (const float*)d_in[13];
  const int*   codes  = (const int*)d_in[14];
  const int*   lens   = (const int*)d_in[15];

  char* ws = (char*)d_ws;
  float*    Mtmp    = (float*)(ws);                         // 256 KB
  _Float16* Mpk     = (_Float16*)(ws + 262144);             // 128 KB
  _Float16* W1pk    = (_Float16*)(ws + 393216);             // 704 KB
  _Float16* Xg      = (_Float16*)(ws + 1114112);            // 1408 KB
  float*    loss_ws = (float*)(ws + 2555904);               // 8 KB

  prep_m<<<DIM, 256, 0, stream>>>(Wq, Wk, Mtmp);
  pack_m<<<128, 256, 0, stream>>>(Mtmp, Mpk);
  pack_w1<<<11 * 64, 256, 0, stream>>>(W1, W1pk);
  fused_sample<<<NSAMP, 256, 0, stream>>>(E, Mpk, Wv, demo, htime, etime,
                                          codes, lens, c_attn, c_pool, Xg);
  mlp_loss<<<NSAMP / 16, 256, 0, stream>>>(Xg, W1pk, b1, W2, b2, labels, loss_ws);
  reduce_loss<<<1, 256, 0, stream>>>(loss_ws, c_attn, c_pool, (float*)d_out);
}
